// GATE9a_53111565582528
// MI455X (gfx1250) — compile-verified, matrix operand order checked by reading, not run on hardware
//
#include <hip/hip_runtime.h>
#include <math.h>

typedef __attribute__((ext_vector_type(16))) __bf16 v16bf;
typedef __attribute__((ext_vector_type(8)))  float  v8f;

static constexpr int   cN = 50000;
static constexpr int   cE = 320000;
static constexpr int   cG = 256;
static constexpr float BN_INV = 0.99999500003749974f;  // 1/sqrt(1+1e-5)

#define TM 128
#define TN 64
#define TK 32

// ---------------------------------------------------------------------------
// Generic GEMM:  C[M x Nc] = act(A[M x K] @ W[K x Nc] + bias)
// fp32 in global; staged to LDS as bf16; fp32 accumulate via
// v_wmma_f32_16x16x32_bf16. Block = 256 threads = 8 waves; tile 128x64;
// each wave owns one 16-row slab x 64 cols -> 4 WMMAs/K-step, A reused 4x.
// B staged transposed so both fragments are two 16B chunks (ds_load_b128).
// Ragged K handled branchlessly: A reads clamped (garbage), B rows >= K are
// selected to zero (cndmask) so the garbage contributes nothing.
// Requires: Nc % 64 == 0, K % 4 == 0 (true for all GEMMs in this net).
// ---------------------------------------------------------------------------
__global__ __launch_bounds__(256)
void k_gemm(const float* __restrict__ A, const float* __restrict__ W,
            const float* __restrict__ bias, float* __restrict__ C,
            int M, int K, int Nc, int relu_act)
{
  __shared__ __bf16 sA [TM][TK];   // (m,k) 128x32 = 8KB
  __shared__ __bf16 sBt[TN][TK];   // (n,k)  64x32 = 4KB

  const int tid  = threadIdx.x;
  const int lane = tid & 31;
  const int wv   = tid >> 5;                 // 0..7 : 16-row slab
  const int m0   = blockIdx.x * TM;
  const int n0   = blockIdx.y * TN;

  const int l15  = lane & 15;
  const int klo  = (lane < 16) ? 0 : 8;      // CDNA5 16-bit A/B K-pair layout
  const int mrow = wv * 16 + l15;

  v8f acc[4] = {};

  for (int k0 = 0; k0 < K; k0 += TK) {
    // ---- stage A: 128x32, 4x float4 per thread, fully branchless ----------
#pragma unroll
    for (int it = 0; it < 4; ++it) {
      int i4 = tid + it * 256;               // 1024 float4 total
      int r  = i4 >> 3;                      // 8 float4 per 32-wide row
      int c  = (i4 & 7) << 2;
      int gm = m0 + r; if (gm >= M) gm = M - 1;   // clamp: rows never stored
      int gk = k0 + c; if (gk >= K) gk = 0;       // clamp: B rows >= K are zero
      float4 v = *(const float4*)(A + (size_t)gm * K + gk);
      union { __bf16 h[4]; uint2 u; } t;
      t.h[0] = (__bf16)v.x; t.h[1] = (__bf16)v.y;
      t.h[2] = (__bf16)v.z; t.h[3] = (__bf16)v.w;
      *(uint2*)&sA[r][c] = t.u;
    }
    // ---- stage B transposed: 32(k) x 64(n), branchless zero-select --------
#pragma unroll
    for (int it = 0; it < 2; ++it) {
      int i4 = tid + it * 256;               // 512 float4 total
      int r  = i4 >> 4;                      // k row: 16 float4 per 64-wide row
      int c  = (i4 & 15) << 2;               // n col
      int gk  = k0 + r;
      bool ok = gk < K;
      int gkc = ok ? gk : (K - 1);           // valid, finite weight row
      float4 v = *(const float4*)(W + (size_t)gkc * Nc + (n0 + c));
      sBt[c + 0][r] = (__bf16)(ok ? v.x : 0.f);
      sBt[c + 1][r] = (__bf16)(ok ? v.y : 0.f);
      sBt[c + 2][r] = (__bf16)(ok ? v.z : 0.f);
      sBt[c + 3][r] = (__bf16)(ok ? v.w : 0.f);
    }
    if (k0 + TK < K)
      __builtin_prefetch((const void*)(A + (size_t)(m0 + mrow) * K + (k0 + TK)), 0, 1);
    __syncthreads();

    // ---- fragments: two contiguous 16B chunks per lane --------------------
    union F { v16bf v; uint4 q[2]; };
    F fa;
    fa.q[0] = *(const uint4*)&sA[mrow][klo];        // K klo..klo+7
    fa.q[1] = *(const uint4*)&sA[mrow][klo + 16];   // K klo+16..klo+23
#pragma unroll
    for (int nb = 0; nb < 4; ++nb) {
      const int ncol = nb * 16 + l15;
      F fb;
      fb.q[0] = *(const uint4*)&sBt[ncol][klo];
      fb.q[1] = *(const uint4*)&sBt[ncol][klo + 16];
      acc[nb] = __builtin_amdgcn_wmma_f32_16x16x32_bf16(false, fa.v, false, fb.v,
                                                        (short)0, acc[nb],
                                                        false, false);
    }
    __syncthreads();
  }

  // C/D layout: VGPR r -> M = r (lanes 0-15) / 8+r (lanes 16-31); N = lane&15
  const int mb = m0 + wv * 16 + ((lane < 16) ? 0 : 8);
#pragma unroll
  for (int nb = 0; nb < 4; ++nb) {
    const int gn = n0 + nb * 16 + l15;
    const float bv = bias ? bias[gn] : 0.f;
#pragma unroll
    for (int r = 0; r < 8; ++r) {
      int gm = mb + r;
      if (gm < M) {
        float v = acc[nb][r] + bv;
        if (relu_act) v = fmaxf(v, 0.f);
        C[(size_t)gm * Nc + gn] = v;
      }
    }
  }
}

// ---------------------------------------------------------------------------
// Element-wise / scatter kernels (float4-vectorized; all dims % 4 == 0)
// ---------------------------------------------------------------------------
__device__ inline float lrelu(float m) { return (m > 0.f) ? m : 0.2f * m; }

__global__ void k_fill4(float* __restrict__ p, float v, size_t n4)
{
  size_t i = (size_t)blockIdx.x * blockDim.x + threadIdx.x;
  if (i < n4) ((float4*)p)[i] = make_float4(v, v, v, v);
}

// cat[e] = [ x[src[e]] (D) | x[dst[e]] (D) | ea[e] (De) ]
__global__ void k_edge_concat(const float* __restrict__ x, const float* __restrict__ ea,
                              const int* __restrict__ src, const int* __restrict__ dst,
                              float* __restrict__ out, int E_, int D, int De)
{
  const int totw  = 2 * D + De;
  const int totw4 = totw >> 2;
  size_t i = (size_t)blockIdx.x * blockDim.x + threadIdx.x;
  size_t n = (size_t)E_ * totw4;
  if (i >= n) return;
  int e = (int)(i / totw4);
  int c = (int)(i % totw4) << 2;
  float4 v;
  if (c < D)           v = *(const float4*)(x + (size_t)src[e] * D + c);
  else if (c < 2 * D)  v = *(const float4*)(x + (size_t)dst[e] * D + (c - D));
  else                 v = *(const float4*)(ea + (size_t)e * De + (c - 2 * D));
  *(float4*)(out + (size_t)e * totw + c) = v;
}

__global__ void k_scatter_mean_acc(const float* __restrict__ vals, const int* __restrict__ dst,
                                   float* __restrict__ out, float* __restrict__ deg,
                                   int E_, int D)
{
  const int D4 = D >> 2;
  size_t i = (size_t)blockIdx.x * blockDim.x + threadIdx.x;
  size_t n = (size_t)E_ * D4;
  if (i >= n) return;
  int e = (int)(i / D4);
  int c = (int)(i % D4) << 2;
  int d = dst[e];
  float4 v = *(const float4*)(vals + (size_t)e * D + c);
  float* o = out + (size_t)d * D + c;
  atomicAdd(o + 0, v.x); atomicAdd(o + 1, v.y);
  atomicAdd(o + 2, v.z); atomicAdd(o + 3, v.w);
  if (c == 0) atomicAdd(&deg[d], 1.f);
}

__global__ void k_div_deg(float* __restrict__ out, const float* __restrict__ deg,
                          int Nn, int D)
{
  const int D4 = D >> 2;
  size_t i = (size_t)blockIdx.x * blockDim.x + threadIdx.x;
  size_t n = (size_t)Nn * D4;
  if (i >= n) return;
  int r = (int)(i / D4);
  float inv = 1.f / fmaxf(deg[r], 1.f);
  float4 v = ((float4*)out)[i];
  v.x *= inv; v.y *= inv; v.z *= inv; v.w *= inv;
  ((float4*)out)[i] = v;
}

__device__ inline void atomicMaxFloat(float* addr, float val)
{
  int* ai  = (int*)addr;
  int  old = *ai;
  while (__int_as_float(old) < val) {
    int assumed = old;
    old = atomicCAS(ai, assumed, __float_as_int(val));
    if (old == assumed) break;
  }
}

// per (edge,h): a = sum_c leaky(xl[s]+xr[d]+ee) * att[h][c]; segment-max into amax
__global__ void k_att_logits(const float* __restrict__ xl, const float* __restrict__ xr,
                             const float* __restrict__ ee, const int* __restrict__ src,
                             const int* __restrict__ dst, const float* __restrict__ att,
                             float* __restrict__ alog, float* __restrict__ amax,
                             int E_, int Nn)
{
  size_t tot = (size_t)(E_ + Nn) * 4;
  size_t i = (size_t)blockIdx.x * blockDim.x + threadIdx.x;
  if (i >= tot) return;
  int h = (int)(i & 3);
  size_t e = i >> 2;
  int s, d;
  if (e < (size_t)E_) { s = src[e]; d = dst[e]; }
  else                { s = d = (int)(e - (size_t)E_); }
  const float4* pl = (const float4*)(xl + (size_t)s * 256 + h * 64);
  const float4* pr = (const float4*)(xr + (size_t)d * 256 + h * 64);
  const float4* pe = (const float4*)(ee + e * 256 + h * 64);
  const float4* pa = (const float4*)(att + h * 64);
  float a = 0.f;
#pragma unroll 4
  for (int c4 = 0; c4 < 16; ++c4) {
    float4 vl = pl[c4], vr = pr[c4], ve = pe[c4], va = pa[c4];
    a += lrelu(vl.x + vr.x + ve.x) * va.x;
    a += lrelu(vl.y + vr.y + ve.y) * va.y;
    a += lrelu(vl.z + vr.z + ve.z) * va.z;
    a += lrelu(vl.w + vr.w + ve.w) * va.w;
  }
  alog[e * 4 + h] = a;
  atomicMaxFloat(&amax[(size_t)d * 4 + h], a);
}

__global__ void k_att_exp(float* __restrict__ alog, const float* __restrict__ amax,
                          float* __restrict__ den, const int* __restrict__ dst,
                          int E_, int Nn)
{
  size_t tot = (size_t)(E_ + Nn) * 4;
  size_t i = (size_t)blockIdx.x * blockDim.x + threadIdx.x;
  if (i >= tot) return;
  int h = (int)(i & 3);
  size_t e = i >> 2;
  int d = (e < (size_t)E_) ? dst[e] : (int)(e - (size_t)E_);
  float ex = __expf(alog[i] - amax[(size_t)d * 4 + h]);
  alog[i] = ex;
  atomicAdd(&den[(size_t)d * 4 + h], ex);
}

__global__ void k_att_scatter(const float* __restrict__ alog, const float* __restrict__ den,
                              const float* __restrict__ xl, const int* __restrict__ src,
                              const int* __restrict__ dst, float* __restrict__ xn,
                              int E_, int Nn)
{
  size_t tot = (size_t)(E_ + Nn) * 64;        // 256 feats / 4
  size_t i = (size_t)blockIdx.x * blockDim.x + threadIdx.x;
  if (i >= tot) return;
  int f = (int)(i & 63) << 2;                 // 4 feats, same head (64-aligned)
  size_t e = i >> 6;
  int h = f >> 6;
  int s, d;
  if (e < (size_t)E_) { s = src[e]; d = dst[e]; }
  else                { s = d = (int)(e - (size_t)E_); }
  float w = alog[e * 4 + h] / den[(size_t)d * 4 + h];
  float4 v = *(const float4*)(xl + (size_t)s * 256 + f);
  float* o = xn + (size_t)d * 256 + f;
  atomicAdd(o + 0, v.x * w); atomicAdd(o + 1, v.y * w);
  atomicAdd(o + 2, v.z * w); atomicAdd(o + 3, v.w * w);
}

__global__ void k_bias_relu(float* __restrict__ x, const float* __restrict__ bias,
                            size_t rows, int D)
{
  const int D4 = D >> 2;
  size_t i = (size_t)blockIdx.x * blockDim.x + threadIdx.x;
  size_t n = rows * D4;
  if (i >= n) return;
  int c = (int)(i % D4) << 2;
  float4 v = ((float4*)x)[i];
  float4 b = *(const float4*)(bias + c);
  v.x = fmaxf(v.x + b.x, 0.f); v.y = fmaxf(v.y + b.y, 0.f);
  v.z = fmaxf(v.z + b.z, 0.f); v.w = fmaxf(v.w + b.w, 0.f);
  ((float4*)x)[i] = v;
}

__global__ void k_pool(const float* __restrict__ x, const int* __restrict__ batch,
                       float* __restrict__ pooled, int Nn, int D)
{
  const int D4 = D >> 2;
  size_t i = (size_t)blockIdx.x * blockDim.x + threadIdx.x;
  size_t n = (size_t)Nn * D4;
  if (i >= n) return;
  int r = (int)(i / D4);
  int c = (int)(i % D4) << 2;
  float4 v = ((const float4*)x)[i];
  float* o = pooled + (size_t)batch[r] * D + c;
  atomicAdd(o + 0, v.x); atomicAdd(o + 1, v.y);
  atomicAdd(o + 2, v.z); atomicAdd(o + 3, v.w);
}

__global__ void k_concat2(const float* __restrict__ u, const float* __restrict__ p,
                          float* __restrict__ out, int Gn, int D)
{
  const int W4 = (2 * D) >> 2;
  size_t i = (size_t)blockIdx.x * blockDim.x + threadIdx.x;
  size_t n = (size_t)Gn * W4;
  if (i >= n) return;
  int g = (int)(i / W4);
  int c = (int)(i % W4) << 2;
  float4 v;
  if (c < D) v = u ? *(const float4*)(u + (size_t)g * D + c) : make_float4(0.f, 0.f, 0.f, 0.f);
  else       v = *(const float4*)(p + (size_t)g * D + (c - D));
  ((float4*)out)[i] = v;
}

__global__ void k_bn(const float* __restrict__ x, const float* __restrict__ g,
                     const float* __restrict__ b, float* __restrict__ o,
                     size_t rows, int D)
{
  const int D4 = D >> 2;
  size_t i = (size_t)blockIdx.x * blockDim.x + threadIdx.x;
  size_t n = rows * D4;
  if (i >= n) return;
  int c = (int)(i % D4) << 2;
  float4 v = ((const float4*)x)[i];
  float4 gg = *(const float4*)(g + c);
  float4 bb = *(const float4*)(b + c);
  v.x = gg.x * (v.x * BN_INV) + bb.x; v.y = gg.y * (v.y * BN_INV) + bb.y;
  v.z = gg.z * (v.z * BN_INV) + bb.z; v.w = gg.w * (v.w * BN_INV) + bb.w;
  ((float4*)o)[i] = v;
}

__global__ void k_fc2(const float* __restrict__ h, const float* __restrict__ w,
                      const float* __restrict__ b, float* __restrict__ out, int Gn)
{
  int g = blockIdx.x * blockDim.x + threadIdx.x;
  if (g >= Gn) return;
  float s = b[0];
  const float4* ph = (const float4*)(h + (size_t)g * 64);
  const float4* pw = (const float4*)w;
#pragma unroll 4
  for (int c4 = 0; c4 < 16; ++c4) {
    float4 hv = ph[c4], wv = pw[c4];
    s += hv.x * wv.x + hv.y * wv.y + hv.z * wv.z + hv.w * wv.w;
  }
  out[g] = s;
}

// ---------------------------------------------------------------------------
// Host-side helpers
// ---------------------------------------------------------------------------
static inline unsigned gblk(size_t n) { return (unsigned)((n + 255) / 256); }

static void launch_gemm(const float* A, const float* W, const float* b, float* C,
                        int M, int K, int Nc, int act, hipStream_t s)
{
  dim3 grid((unsigned)((M + TM - 1) / TM), (unsigned)(Nc / TN));
  k_gemm<<<grid, 256, 0, s>>>(A, W, b, C, M, K, Nc, act);
}

// One MetaLayer: EdgeModel MLP -> GATv2(+ReLU) -> GlobalModel MLP
static void run_meta(const float* xin, int Din, const float* eain, int Dein,
                     const float* uin,
                     const int* src, const int* dst, const int* batch,
                     const float* ew1, const float* eb1, int Eh,
                     const float* ew2, const float* eb2, int Deo,
                     const float* lw, const float* lb,
                     const float* rw, const float* rb,
                     const float* eww, const float* att, const float* gbias,
                     const float* uw1, const float* ub1,
                     const float* uw2, const float* ub2,
                     float* cat, float* hid, float* eaAll, float* deg,
                     float* xl, float* xr, float* ee,
                     float* alog, float* amax, float* den,
                     float* xn, float* pooled, float* ucat, float* uh, float* uout,
                     hipStream_t st)
{
  const int catW = 2 * Din + Dein;
  // EdgeModel
  k_edge_concat<<<gblk((size_t)cE * (catW >> 2)), 256, 0, st>>>(xin, eain, src, dst, cat, cE, Din, Dein);
  launch_gemm(cat, ew1, eb1, hid, cE, catW, Eh, 1, st);
  launch_gemm(hid, ew2, eb2, eaAll, cE, Eh, Deo, 0, st);
  // self-loop edge features = scatter-mean over dst
  k_fill4<<<gblk((size_t)cN >> 2), 256, 0, st>>>(deg, 0.f, (size_t)cN >> 2);
  k_fill4<<<gblk(((size_t)cN * Deo) >> 2), 256, 0, st>>>(eaAll + (size_t)cE * Deo, 0.f, ((size_t)cN * Deo) >> 2);
  k_scatter_mean_acc<<<gblk((size_t)cE * (Deo >> 2)), 256, 0, st>>>(eaAll, dst, eaAll + (size_t)cE * Deo, deg, cE, Deo);
  k_div_deg<<<gblk((size_t)cN * (Deo >> 2)), 256, 0, st>>>(eaAll + (size_t)cE * Deo, deg, cN, Deo);
  // GATv2 projections (WMMA GEMMs)
  launch_gemm(xin, lw, lb, xl, cN, Din, 256, 0, st);
  launch_gemm(xin, rw, rb, xr, cN, Din, 256, 0, st);
  launch_gemm(eaAll, eww, nullptr, ee, cE + cN, Deo, 256, 0, st);
  // scatter softmax attention
  k_fill4<<<gblk((size_t)cN), 256, 0, st>>>(amax, -1e30f, (size_t)cN);
  k_fill4<<<gblk((size_t)cN), 256, 0, st>>>(den, 0.f, (size_t)cN);
  k_fill4<<<gblk(((size_t)cN * 256) >> 2), 256, 0, st>>>(xn, 0.f, ((size_t)cN * 256) >> 2);
  k_att_logits<<<gblk((size_t)(cE + cN) * 4), 256, 0, st>>>(xl, xr, ee, src, dst, att, alog, amax, cE, cN);
  k_att_exp<<<gblk((size_t)(cE + cN) * 4), 256, 0, st>>>(alog, amax, den, dst, cE, cN);
  k_att_scatter<<<gblk((size_t)(cE + cN) * 64), 256, 0, st>>>(alog, den, xl, src, dst, xn, cE, cN);
  k_bias_relu<<<gblk((size_t)cN * 64), 256, 0, st>>>(xn, gbias, (size_t)cN, 256);
  // GlobalModel
  k_fill4<<<gblk(((size_t)cG * 256) >> 2), 256, 0, st>>>(pooled, 0.f, ((size_t)cG * 256) >> 2);
  k_pool<<<gblk((size_t)cN * 64), 256, 0, st>>>(xn, batch, pooled, cN, 256);
  k_concat2<<<gblk((size_t)cG * 128), 256, 0, st>>>(uin, pooled, ucat, cG, 256);
  launch_gemm(ucat, uw1, ub1, uh, cG, 512, 256, 1, st);
  launch_gemm(uh, uw2, ub2, uout, cG, 256, 256, 0, st);
}

// ---------------------------------------------------------------------------
extern "C" void kernel_launch(void* const* d_in, const int* in_sizes, int n_in,
                              void* d_out, int out_size, void* d_ws, size_t ws_size,
                              hipStream_t stream)
{
  (void)in_sizes; (void)n_in; (void)out_size; (void)ws_size;

  const float* x     = (const float*)d_in[0];
  const float* ea    = (const float*)d_in[1];
  const float* e1_w1 = (const float*)d_in[2];
  const float* e1_b1 = (const float*)d_in[3];
  const float* e1_w2 = (const float*)d_in[4];
  const float* e1_b2 = (const float*)d_in[5];
  const float* g1_lw = (const float*)d_in[6];
  const float* g1_lb = (const float*)d_in[7];
  const float* g1_rw = (const float*)d_in[8];
  const float* g1_rb = (const float*)d_in[9];
  const float* g1_ew = (const float*)d_in[10];
  const float* g1_att  = (const float*)d_in[11];
  const float* g1_bias = (const float*)d_in[12];
  const float* u1_w1 = (const float*)d_in[13];
  const float* u1_b1 = (const float*)d_in[14];
  const float* u1_w2 = (const float*)d_in[15];
  const float* u1_b2 = (const float*)d_in[16];
  const float* bn_n_g = (const float*)d_in[17];
  const float* bn_n_b = (const float*)d_in[18];
  const float* bn_e_g = (const float*)d_in[19];
  const float* bn_e_b = (const float*)d_in[20];
  const float* bn_u_g = (const float*)d_in[21];
  const float* bn_u_b = (const float*)d_in[22];
  const float* e2_w1 = (const float*)d_in[23];
  const float* e2_b1 = (const float*)d_in[24];
  const float* e2_w2 = (const float*)d_in[25];
  const float* e2_b2 = (const float*)d_in[26];
  const float* g2_lw = (const float*)d_in[27];
  const float* g2_lb = (const float*)d_in[28];
  const float* g2_rw = (const float*)d_in[29];
  const float* g2_rb = (const float*)d_in[30];
  const float* g2_ew = (const float*)d_in[31];
  const float* g2_att  = (const float*)d_in[32];
  const float* g2_bias = (const float*)d_in[33];
  const float* u2_w1 = (const float*)d_in[34];
  const float* u2_b1 = (const float*)d_in[35];
  const float* u2_w2 = (const float*)d_in[36];
  const float* u2_b2 = (const float*)d_in[37];
  const float* fc1_w = (const float*)d_in[38];
  const float* fc1_b = (const float*)d_in[39];
  const float* fc2_w = (const float*)d_in[40];
  const float* fc2_b = (const float*)d_in[41];
  const int* edge_index = (const int*)d_in[42];
  const int* batch      = (const int*)d_in[43];
  const int* src = edge_index;
  const int* dst = edge_index + cE;

  float* ws  = (float*)d_ws;
  size_t off = 0;
  auto carve = [&](size_t n) { float* p = ws + off; off += n; return p; };

  // ---- layer 1 scratch ----
  float* cat1   = carve((size_t)cE * 144);
  float* hid1   = carve((size_t)cE * 64);
  float* eaAll1 = carve((size_t)(cE + cN) * 128);   // [ea2 | loop_ea]
  float* deg    = carve((size_t)cN);
  float* xl1    = carve((size_t)cN * 256);
  float* xr1    = carve((size_t)cN * 256);
  float* ee1    = carve((size_t)(cE + cN) * 256);
  float* alog1  = carve((size_t)(cE + cN) * 4);
  float* amax1  = carve((size_t)cN * 4);
  float* den1   = carve((size_t)cN * 4);
  float* xn1    = carve((size_t)cN * 256);
  float* pool1  = carve((size_t)cG * 256);
  float* ucat1  = carve((size_t)cG * 512);
  float* uh1    = carve((size_t)cG * 256);
  float* u1     = carve((size_t)cG * 256);
  float* x1     = carve((size_t)cN * 256);
  float* ea1bn  = carve((size_t)cE * 128);
  float* u1bn   = carve((size_t)cG * 256);
  // ---- layer 2 scratch ----
  float* cat2   = carve((size_t)cE * 640);
  float* hid2   = carve((size_t)cE * 128);
  float* eaAll2 = carve((size_t)(cE + cN) * 128);
  float* xl2    = carve((size_t)cN * 256);
  float* xr2    = carve((size_t)cN * 256);
  float* ee2    = carve((size_t)(cE + cN) * 256);
  float* alog2  = carve((size_t)(cE + cN) * 4);
  float* amax2  = carve((size_t)cN * 4);
  float* den2   = carve((size_t)cN * 4);
  float* xn2    = carve((size_t)cN * 256);
  float* pool2  = carve((size_t)cG * 256);
  float* ucat2  = carve((size_t)cG * 512);
  float* uh2    = carve((size_t)cG * 256);
  float* u2     = carve((size_t)cG * 256);
  float* fch    = carve((size_t)cG * 64);

  // ================= meta layer 1 (u0 = zeros) =================
  run_meta(x, 64, ea, 16, /*uin=*/nullptr, src, dst, batch,
           e1_w1, e1_b1, 64, e1_w2, e1_b2, 128,
           g1_lw, g1_lb, g1_rw, g1_rb, g1_ew, g1_att, g1_bias,
           u1_w1, u1_b1, u1_w2, u1_b2,
           cat1, hid1, eaAll1, deg, xl1, xr1, ee1, alog1, amax1, den1,
           xn1, pool1, ucat1, uh1, u1, stream);

  // BatchNorm (eval mode, running stats 0/1)
  k_bn<<<gblk((size_t)cN * 64), 256, 0, stream>>>(xn1, bn_n_g, bn_n_b, x1, (size_t)cN, 256);
  k_bn<<<gblk((size_t)cE * 32), 256, 0, stream>>>(eaAll1, bn_e_g, bn_e_b, ea1bn, (size_t)cE, 128);
  k_bn<<<gblk((size_t)cG * 64), 256, 0, stream>>>(u1, bn_u_g, bn_u_b, u1bn, (size_t)cG, 256);

  // ================= meta layer 2 =================
  run_meta(x1, 256, ea1bn, 128, u1bn, src, dst, batch,
           e2_w1, e2_b1, 128, e2_w2, e2_b2, 128,
           g2_lw, g2_lb, g2_rw, g2_rb, g2_ew, g2_att, g2_bias,
           u2_w1, u2_b1, u2_w2, u2_b2,
           cat2, hid2, eaAll2, deg, xl2, xr2, ee2, alog2, amax2, den2,
           xn2, pool2, ucat2, uh2, u2, stream);

  // ================= head =================
  launch_gemm(u2, fc1_w, fc1_b, fch, cG, 256, 64, 1, stream);
  k_fc2<<<gblk((size_t)cG), 256, 0, stream>>>(fch, fc2_w, fc2_b, (float*)d_out, cG);
}
